// LSTM_57277683860017
// MI455X (gfx1250) — matrix operand order repaired and checked by reading, not compiled, not run on hardware
//
#include <hip/hip_runtime.h>

typedef __attribute__((ext_vector_type(16))) _Float16 v16h;
typedef __attribute__((ext_vector_type(8)))  _Float16 v8h;
typedef __attribute__((ext_vector_type(8)))  float    v8f;
typedef __attribute__((ext_vector_type(4)))  float    v4f;

#define B_  64
#define T_  1024
#define F_  512
#define H_  128
#define G_  512   /* 4*H */

union V16U { v16h v; v8h h[2]; };

__device__ __forceinline__ float sigmf(float x) {
  return 1.0f / (1.0f + __expf(-x));
}
__device__ __forceinline__ float tanh_fast(float x) {
  // tanh(x) = 1 - 2/(exp(2x)+1)  -> one v_exp_f32
  return 1.0f - 2.0f / (__expf(2.0f * x) + 1.0f);
}

// ---------------------------------------------------------------------------
// Kernel 1: xw[m][g] = sum_f x[m][f]*W_ih[g][f] + b_ih[g] + b_hh[g]
// M = B*T = 65536, N = 512, K = 512.  f16 WMMA, f32 accumulate.
// Block = 256 threads (8 waves), 64x64 output tile; wave -> 16(M) x 32(N).
// ---------------------------------------------------------------------------
__global__ __launch_bounds__(256) void lstm_xproj(
    const float* __restrict__ x, const float* __restrict__ Wih,
    const float* __restrict__ bih, const float* __restrict__ bhh,
    float* __restrict__ xw)
{
  __shared__ _Float16 As[64][40];  // [m][k], row padded to 40 halves (80B)
  __shared__ _Float16 Bs[32][72];  // [k][n], row padded to 72 halves (144B)

  const int tid  = threadIdx.x;
  const int lane = tid & 31;
  const int wave = tid >> 5;
  const int m0 = blockIdx.x * 64;
  const int n0 = blockIdx.y * 64;

  const int m_off = (wave & 3) * 16;
  const int n_off = (wave >> 2) * 32;

  v8f c0 = {}; v8f c1 = {};

  // A staging map: 4 threads/row, 8 k each
  const int arow = tid >> 2;
  const int acol = (tid & 3) * 8;
  // B staging map: one W row segment -> transposed column of Bs
  const int bn = tid & 63;
  const int bk = (tid >> 6) * 8;

  const int am  = m_off + (lane & 15);
  const int akb = (lane >> 4) * 8;

  for (int k0 = 0; k0 < F_; k0 += 32) {
    { // x tile -> As (f32 -> f16)
      const float* src = x + (size_t)(m0 + arow) * F_ + k0 + acol;
      v4f f0 = *(const v4f*)(src);
      v4f f1 = *(const v4f*)(src + 4);
      #pragma unroll
      for (int i = 0; i < 4; ++i) {
        As[arow][acol + i]     = (_Float16)f0[i];
        As[arow][acol + 4 + i] = (_Float16)f1[i];
      }
    }
    { // W tile -> Bs transposed:  Bs[k][n] = W[n0+n][k0+k]
      const float* src = Wih + (size_t)(n0 + bn) * F_ + k0 + bk;
      v4f f0 = *(const v4f*)(src);
      v4f f1 = *(const v4f*)(src + 4);
      #pragma unroll
      for (int i = 0; i < 4; ++i) {
        Bs[bk + i][bn]     = (_Float16)f0[i];
        Bs[bk + 4 + i][bn] = (_Float16)f1[i];
      }
    }
    __syncthreads();

    // A fragment (ISA layout: m=L%16, k=(L/16)*8 + (j/8)*16 + j%8)
    V16U a;
    a.h[0] = *(const v8h*)&As[am][akb];
    a.h[1] = *(const v8h*)&As[am][akb + 16];
    // B fragments (ISA layout: n=L%16, k=(L/16)*8 + (j/8)*16 + j%8)
    V16U b0, b1;
    #pragma unroll
    for (int i = 0; i < 8; ++i) {
      b0.v[i]     = Bs[akb + i][n_off + (lane & 15)];
      b0.v[8 + i] = Bs[akb + 16 + i][n_off + (lane & 15)];
      b1.v[i]     = Bs[akb + i][n_off + 16 + (lane & 15)];
      b1.v[8 + i] = Bs[akb + 16 + i][n_off + 16 + (lane & 15)];
    }

    c0 = __builtin_amdgcn_wmma_f32_16x16x32_f16(false, a.v, false, b0.v,
                                                (short)0, c0, false, false);
    c1 = __builtin_amdgcn_wmma_f32_16x16x32_f16(false, a.v, false, b1.v,
                                                (short)0, c1, false, false);
    __syncthreads();
  }

  // Epilogue: add combined bias, store f32 (C layout: m=r+(L/16)*8, n=L%16)
  const int nloc0 = n0 + n_off + (lane & 15);
  const int nloc1 = nloc0 + 16;
  const float bias0 = bih[nloc0] + bhh[nloc0];
  const float bias1 = bih[nloc1] + bhh[nloc1];
  const int mbase = m0 + m_off + (lane >> 4) * 8;
  #pragma unroll
  for (int r = 0; r < 8; ++r) {
    xw[(size_t)(mbase + r) * G_ + nloc0] = c0[r] + bias0;
    xw[(size_t)(mbase + r) * G_ + nloc1] = c1[r] + bias1;
  }
}

// ---------------------------------------------------------------------------
// Kernel 2: persistent single-workgroup LSTM recurrence + classifier.
// 1024 threads = 32 waves. Wave w: batch tile m=w%4, hidden tile h=w/4.
// Each wave computes the i/f/g/o gate tiles for its OWN (b,h) 16x16 block,
// so cell state stays in that wave's registers (WMMA C-fragment layout).
// W_hh^T fragments are hoisted into registers (time-invariant, 128 VGPRs);
// only h (A-fragments) travels through LDS each step. Next step's xw seed
// lines are prefetched (global_prefetch_b8) to hide HBM/L2 latency on the
// sequential dependence chain.
// ---------------------------------------------------------------------------
__global__ __launch_bounds__(1024) void lstm_recur(
    const float* __restrict__ xw, const float* __restrict__ Whh,
    const float* __restrict__ Wcls, const float* __restrict__ bcls,
    float* __restrict__ out)
{
  extern __shared__ _Float16 smem[];
  _Float16* Wt = smem;               // [128][512]  Wt[k][n] = Whh[n][k]
  _Float16* hb = smem + H_ * G_;     // [64][128]   current hidden state

  const int tid  = threadIdx.x;
  const int lane = tid & 31;
  const int wave = tid >> 5;
  const int m_tile = wave & 3;       // batch block (16 rows)
  const int h_tile = wave >> 2;      // hidden block (16 cols), 0..7
  const int ln15 = lane & 15;

  // Stage Whh^T (f32 -> f16) into LDS
  for (int idx = tid; idx < G_ * H_; idx += 1024) {
    int n = idx >> 7;      // gate-row index 0..511
    int k = idx & 127;     // hidden index 0..127
    Wt[k * G_ + n] = (_Float16)Whh[idx];
  }
  // h(0) = 0
  for (int idx = tid; idx < B_ * H_; idx += 1024) hb[idx] = (_Float16)0.0f;
  __syncthreads();

  // Hoist all 16 time-invariant B fragments (4 gates x 4 k-chunks) into
  // registers: B layout n=L%16, k=(L/16)*8 + (j/8)*16 + j%8.
  V16U bfr[4][4];
  #pragma unroll
  for (int q = 0; q < 4; ++q) {
    const int ng = q * H_ + h_tile * 16;
    #pragma unroll
    for (int s = 0; s < 4; ++s) {
      #pragma unroll
      for (int i = 0; i < 8; ++i) {
        bfr[q][s].v[i]     = Wt[(32 * s + (lane >> 4) * 8 + i) * G_ + ng + ln15];
        bfr[q][s].v[8 + i] = Wt[(32 * s + (lane >> 4) * 8 + 16 + i) * G_ + ng + ln15];
      }
    }
  }

  v8f cst = {};  // persistent cell state, C-fragment layout

  const int am    = m_tile * 16 + ln15;            // A-frag row (batch)
  const int akb   = (lane >> 4) * 8;               // A-frag k sub-base
  const int cn    = h_tile * 16 + ln15;            // this lane's hidden col
  const int bbase = m_tile * 16 + (lane >> 4) * 8; // C-frag batch row base

  // Per-gate base pointers for the xw accumulator seed:
  //   seed(q,r,t) = xb[q][ r*T*G + t*G ]
  const float* xb[4];
  const float* pf[4];  // prefetch: one 64B row-line per lane (row = am)
  #pragma unroll
  for (int q = 0; q < 4; ++q) {
    const int ng = q * H_ + h_tile * 16;
    xb[q] = xw + (size_t)bbase * T_ * G_ + ng + ln15;
    pf[q] = xw + (size_t)am * T_ * G_ + ng;
  }

  for (int t = 0; t < T_; ++t) {
    // A fragments for k = 0..127 (4 chunks of 32), from hb
    V16U a[4];
    #pragma unroll
    for (int s = 0; s < 4; ++s) {
      a[s].h[0] = *(const v8h*)&hb[am * H_ + 32 * s + akb];
      a[s].h[1] = *(const v8h*)&hb[am * H_ + 32 * s + akb + 16];
    }
    __syncthreads();  // everyone done reading h(t-1) before it is overwritten

    // Prefetch next step's xw seed lines (speculative, RT)
    const int tn = (t + 1 < T_) ? (t + 1) : t;
    #pragma unroll
    for (int q = 0; q < 4; ++q)
      __builtin_prefetch(pf[q] + (size_t)tn * G_, 0, 3);

    v8f acc[4];
    #pragma unroll
    for (int q = 0; q < 4; ++q) {        // i, f, g, o gate tiles
      v8f cacc;
      #pragma unroll
      for (int r = 0; r < 8; ++r)        // accumulator seeded with xw(:,t,:)
        cacc[r] = xb[q][(size_t)r * T_ * G_ + (size_t)t * G_];
      #pragma unroll
      for (int s = 0; s < 4; ++s)
        cacc = __builtin_amdgcn_wmma_f32_16x16x32_f16(false, a[s].v, false,
                                                      bfr[q][s].v, (short)0,
                                                      cacc, false, false);
      acc[q] = cacc;
    }

    // Gate nonlinearities + state update (all data wave-local)
    #pragma unroll
    for (int r = 0; r < 8; ++r) {
      float ig = sigmf(acc[0][r]);
      float fg = sigmf(acc[1][r]);
      float gg = tanh_fast(acc[2][r]);
      float og = sigmf(acc[3][r]);
      float cnew = fg * cst[r] + ig * gg;
      cst[r] = cnew;
      float hnew = og * tanh_fast(cnew);
      hb[(bbase + r) * H_ + cn] = (_Float16)hnew;
    }
    __syncthreads();  // h(t) visible to all waves
  }

  // Classifier: logits[b] = h_f[b, T-1, :] . W_cls + b_cls
  if (tid < B_) {
    float accv = bcls[0];
    for (int hh = 0; hh < H_; ++hh)
      accv += (float)hb[tid * H_ + hh] * Wcls[hh];
    out[tid] = accv;
  }
}

// ---------------------------------------------------------------------------
extern "C" void kernel_launch(void* const* d_in, const int* in_sizes, int n_in,
                              void* d_out, int out_size, void* d_ws, size_t ws_size,
                              hipStream_t stream) {
  (void)in_sizes; (void)n_in; (void)out_size; (void)ws_size;
  const float* x    = (const float*)d_in[0];
  const float* Wih  = (const float*)d_in[1];
  const float* Whh  = (const float*)d_in[2];
  const float* bih  = (const float*)d_in[3];
  const float* bhh  = (const float*)d_in[4];
  // d_in[5..8] = backward-direction params: dead code in the reference output
  const float* Wcls = (const float*)d_in[9];
  const float* bcls = (const float*)d_in[10];
  float* out = (float*)d_out;
  float* xw  = (float*)d_ws;   // [B*T, 4H] f32 = 128 MiB scratch

  dim3 g1(B_ * T_ / 64, G_ / 64);
  lstm_xproj<<<g1, 256, 0, stream>>>(x, Wih, bih, bhh, xw);

  size_t smem = (size_t)(H_ * G_ + B_ * H_) * sizeof(_Float16);  // 144 KiB
  lstm_recur<<<1, 1024, smem, stream>>>(xw, Whh, Wcls, bcls, out);
}
